// FluxSingleTransformerBlock_28853590294600
// MI455X (gfx1250) — matrix operand order repaired and checked by reading, not compiled, not run on hardware
//
#include <hip/hip_runtime.h>
#include <hip/hip_bf16.h>
#include <math.h>

typedef __attribute__((ext_vector_type(16))) __bf16 v16bf;
typedef __attribute__((ext_vector_type(8)))  __bf16 v8bf;
typedef __attribute__((ext_vector_type(8)))  float  v8f;
typedef __attribute__((ext_vector_type(4)))  int    v4i;

#define DIM    3072
#define NHEAD  24
#define DH     128
#define MLPD   12288
#define SEQ    4096
#define BATCH  2
#define NROWS  (BATCH*SEQ)       // 8192
#define NQKV   (3*DIM)           // 9216
#define N1     (NQKV+MLPD)       // 21504
#define KTOT2  (DIM+MLPD)        // 15360
#define EPSV   1e-6f

// ---------------- workspace layout (bytes) ----------------
#define OFF_EMB  ((size_t)0)                        // 2*9216 f32      = 73728
#define OFF_X    ((size_t)73728)                    // 8192*3072 bf16  = 50331648
#define OFF_W1T  ((size_t)50405376)                 // 21504*3072 bf16 = 132120576
#define OFF_W2T  ((size_t)182525952)                // 3072*15360 bf16 = 94371840
#define OFF_QKV  ((size_t)276897792)                // 8192*9216 bf16  = 150994944
#define OFF_MLP  ((size_t)427892736)                // 8192*12288 bf16 = 201326592
#define OFF_Q    ((size_t)629219328)                // 8192*3072 bf16
#define OFF_K    ((size_t)679550976)
#define OFF_VT   ((size_t)729882624)
#define OFF_ATT  ((size_t)780214272)

// ---------------- CDNA5 async global->LDS path ----------------
#if defined(__has_builtin)
#if __has_builtin(__builtin_amdgcn_global_load_async_to_lds_b128)
#define HAS_ASYNC_LDS 1
#endif
#endif

static __device__ __forceinline__ void async_copy16(const __bf16* g, __bf16* l) {
#ifdef HAS_ASYNC_LDS
    __bf16* gn = const_cast<__bf16*>(g);
    __builtin_amdgcn_global_load_async_to_lds_b128(
        (__attribute__((address_space(1))) v4i*)gn,
        (__attribute__((address_space(3))) v4i*)l, 0, 0);
#else
    *(v8bf*)l = *(const v8bf*)g;
#endif
}

static __device__ __forceinline__ void wait_async_prev_tile() {
#ifdef HAS_ASYNC_LDS
#if __has_builtin(__builtin_amdgcn_s_wait_asynccnt)
    __builtin_amdgcn_s_wait_asynccnt(8);
#else
    asm volatile("s_wait_asynccnt 0x8" ::: "memory");
#endif
#endif
}
static __device__ __forceinline__ void wait_async_all() {
#ifdef HAS_ASYNC_LDS
#if __has_builtin(__builtin_amdgcn_s_wait_asynccnt)
    __builtin_amdgcn_s_wait_asynccnt(0);
#else
    asm volatile("s_wait_asynccnt 0x0" ::: "memory");
#endif
#endif
}

// ---------------- WMMA helpers ----------------
static __device__ __forceinline__ v8f wmma_bf16(v16bf a, v16bf b, v8f c) {
    return __builtin_amdgcn_wmma_f32_16x16x32_bf16(false, a, false, b, (short)0, c, false, false);
}

// A fragment (16x32 bf16): lane holds row M=lane&15; K halves split per lane half.
static __device__ __forceinline__ v16bf load_a_frag(const __bf16* base, size_t stride, int lane) {
    const __bf16* p = base + (size_t)(lane & 15) * stride + ((lane >> 4) * 8);
    v16bf a;
    ((v8bf*)&a)[0] = *(const v8bf*)(p);        // K = half+0..7
    ((v8bf*)&a)[1] = *(const v8bf*)(p + 16);   // K = half+16..23
    return a;
}

// B fragment (32x16 bf16): lane holds column N=lane&15; K = half+0..7 and half+16..23 (half = 8*(lane>>4)).
static __device__ __forceinline__ v16bf load_b_frag(const __bf16* base, size_t stride, int lane) {
    const __bf16* p = base + (size_t)(lane & 15) * stride + ((lane >> 4) * 8);
    v16bf b;
    ((v8bf*)&b)[0] = *(const v8bf*)(p);
    ((v8bf*)&b)[1] = *(const v8bf*)(p + 16);
    return b;
}

static __device__ __forceinline__ float gelu_tanh(float x) {
    float x3 = x * x * x;
    return 0.5f * x * (1.0f + tanhf(0.7978845608028654f * (x + 0.044715f * x3)));
}

// ---------------- weight convert / transpose ----------------
__global__ void convert_w1t_kernel(const float* __restrict__ w1, __bf16* __restrict__ w1t) {
    const size_t n = (size_t)N1 * DIM;
    for (size_t i = (size_t)blockIdx.x * 256 + threadIdx.x; i < n; i += (size_t)gridDim.x * 256) {
        size_t col = i / DIM, k = i % DIM;
        w1t[i] = (__bf16)w1[k * N1 + col];
    }
}
__global__ void convert_w2t_kernel(const float* __restrict__ w2, __bf16* __restrict__ w2t) {
    const size_t n = (size_t)DIM * KTOT2;
    for (size_t i = (size_t)blockIdx.x * 256 + threadIdx.x; i < n; i += (size_t)gridDim.x * 256) {
        size_t col = i / KTOT2, k = i % KTOT2;
        w2t[i] = (__bf16)w2[k * DIM + col];
    }
}

// ---------------- emb = silu(temb) @ w_lin + b_lin ----------------
__global__ __launch_bounds__(256) void emb_kernel(const float* __restrict__ temb,
                                                  const float* __restrict__ w_lin,
                                                  const float* __restrict__ b_lin,
                                                  float* __restrict__ emb) {
    __shared__ float st[DIM];
    const int b = blockIdx.y;
    for (int i = threadIdx.x; i < DIM; i += 256) {
        float t = temb[b * DIM + i];
        st[i] = t / (1.0f + __expf(-t));
    }
    __syncthreads();
    const int c = blockIdx.x * 256 + threadIdx.x;
    float acc = b_lin[c];
    for (int i = 0; i < DIM; i++) acc += st[i] * w_lin[(size_t)i * NQKV + c];
    emb[(size_t)b * NQKV + c] = acc;
}

// ---------------- LayerNorm + modulate -> x (bf16) ----------------
__global__ __launch_bounds__(256) void lnmod_kernel(const float* __restrict__ hs,
                                                    const float* __restrict__ emb,
                                                    __bf16* __restrict__ x) {
    const int row = blockIdx.x;
    const int b = row >> 12;
    const float* h = hs + (size_t)row * DIM;
    float lv[12], s = 0.f, ss = 0.f;
#pragma unroll
    for (int i = 0; i < 12; i++) {
        float v = h[threadIdx.x + i * 256];
        lv[i] = v; s += v; ss += v * v;
    }
#pragma unroll
    for (int m = 1; m < 32; m <<= 1) { s += __shfl_xor(s, m, 32); ss += __shfl_xor(ss, m, 32); }
    __shared__ float sh[16];
    const int wid = threadIdx.x >> 5, lane = threadIdx.x & 31;
    if (lane == 0) { sh[wid] = s; sh[8 + wid] = ss; }
    __syncthreads();
    float st = 0.f, sst = 0.f;
#pragma unroll
    for (int w = 0; w < 8; w++) { st += sh[w]; sst += sh[8 + w]; }
    const float mu = st / (float)DIM;
    const float var = sst / (float)DIM - mu * mu;
    const float rs = rsqrtf(var + EPSV);
    const float* shf = emb + (size_t)b * NQKV;
    const float* scl = shf + DIM;
#pragma unroll
    for (int i = 0; i < 12; i++) {
        int c = threadIdx.x + i * 256;
        float v = (lv[i] - mu) * rs * (1.0f + scl[c]) + shf[c];
        x[(size_t)row * DIM + c] = (__bf16)v;
    }
}

// ---------------- GEMM1: x(8192x3072) @ w1t^T -> qkv(bf16) + gelu(mlp)(bf16) ----------------
// Double-buffered BK=64 tiles staged with async global->LDS copies (ASYNCcnt pipeline).
__global__ __launch_bounds__(256) void gemm1_kernel(const __bf16* __restrict__ X,
                                                    const __bf16* __restrict__ WT,
                                                    const float* __restrict__ bias,
                                                    __bf16* __restrict__ qkv,
                                                    __bf16* __restrict__ mlp) {
    __shared__ __align__(64) __bf16 As[2][128 * 64];
    __shared__ __align__(64) __bf16 Bs[2][128 * 64];
    const int tid = threadIdx.x, lane = tid & 31, wid = tid >> 5;
    const int wm = wid >> 1, wn = wid & 1;
    const int row0 = blockIdx.y * 128, col0 = blockIdx.x * 128;
    v8f acc[2][4];
#pragma unroll
    for (int i = 0; i < 2; i++)
#pragma unroll
        for (int j = 0; j < 4; j++)
#pragma unroll
            for (int r = 0; r < 8; r++) acc[i][j][r] = 0.f;

    const int nkt = DIM / 64;   // 48

    // issue tile 0
#pragma unroll
    for (int t = 0; t < 4; t++) {
        int ci = tid + t * 256;            // 0..1023
        int rr = ci >> 3;                  // 0..127
        int cc = (ci & 7) * 8;             // 0..56
        async_copy16(&X[(size_t)(row0 + rr) * DIM + cc], &As[0][rr * 64 + cc]);
        async_copy16(&WT[(size_t)(col0 + rr) * DIM + cc], &Bs[0][rr * 64 + cc]);
    }

    for (int kt = 0; kt < nkt; kt++) {
        const int cur = kt & 1;
        if (kt + 1 < nkt) {
            const int nxt = cur ^ 1;
            const int k0 = (kt + 1) * 64;
#pragma unroll
            for (int t = 0; t < 4; t++) {
                int ci = tid + t * 256;
                int rr = ci >> 3;
                int cc = (ci & 7) * 8;
                async_copy16(&X[(size_t)(row0 + rr) * DIM + k0 + cc], &As[nxt][rr * 64 + cc]);
                async_copy16(&WT[(size_t)(col0 + rr) * DIM + k0 + cc], &Bs[nxt][rr * 64 + cc]);
            }
            wait_async_prev_tile();
        } else {
            wait_async_all();
        }
        __syncthreads();
#pragma unroll
        for (int kc = 0; kc < 2; kc++) {
            v16bf a[2], bb[4];
#pragma unroll
            for (int i = 0; i < 2; i++)
                a[i] = load_a_frag(&As[cur][(wm * 32 + i * 16) * 64 + kc * 32], 64, lane);
#pragma unroll
            for (int j = 0; j < 4; j++)
                bb[j] = load_b_frag(&Bs[cur][(wn * 64 + j * 16) * 64 + kc * 32], 64, lane);
#pragma unroll
            for (int i = 0; i < 2; i++)
#pragma unroll
                for (int j = 0; j < 4; j++)
                    acc[i][j] = wmma_bf16(a[i], bb[j], acc[i][j]);
        }
        __syncthreads();
    }
#pragma unroll
    for (int i = 0; i < 2; i++)
#pragma unroll
        for (int j = 0; j < 4; j++)
#pragma unroll
            for (int r = 0; r < 8; r++) {
                int m = wm * 32 + i * 16 + r + 8 * (lane >> 4);
                int n = wn * 64 + j * 16 + (lane & 15);
                int rg = row0 + m, cg = col0 + n;
                float v = acc[i][j][r] + bias[cg];
                if (cg < NQKV) qkv[(size_t)rg * NQKV + cg] = (__bf16)v;
                else mlp[(size_t)rg * MLPD + (cg - NQKV)] = (__bf16)gelu_tanh(v);
            }
}

// ---------------- RMS + RoPE + head reorder; V transposed ----------------
__global__ __launch_bounds__(256) void rmsrope_kernel(const __bf16* __restrict__ qkv,
                                                      const float* __restrict__ rope,
                                                      const float* __restrict__ qs,
                                                      const float* __restrict__ ks,
                                                      __bf16* __restrict__ qb,
                                                      __bf16* __restrict__ kb,
                                                      __bf16* __restrict__ vt) {
    const int wid = threadIdx.x >> 5, lane = threadIdx.x & 31;
    const size_t widg = (size_t)blockIdx.x * 8 + wid;   // row in (B,L,3,H)
    const int h = (int)(widg % NHEAD);
    const int t3 = (int)((widg / NHEAD) % 3);
    const int l = (int)((widg / (NHEAD * 3)) % SEQ);
    const int b = (int)(widg / ((size_t)NHEAD * 3 * SEQ));
    const int d0 = lane * 4;
    const __bf16* src = qkv + widg * DH + d0;
    float x0 = src[0], x1 = src[1], x2 = src[2], x3 = src[3];
    if (t3 == 2) {
        const size_t base = ((size_t)(b * NHEAD + h) * DH);
        vt[(base + d0 + 0) * SEQ + l] = (__bf16)x0;
        vt[(base + d0 + 1) * SEQ + l] = (__bf16)x1;
        vt[(base + d0 + 2) * SEQ + l] = (__bf16)x2;
        vt[(base + d0 + 3) * SEQ + l] = (__bf16)x3;
    } else {
        float ssum = x0 * x0 + x1 * x1 + x2 * x2 + x3 * x3;
#pragma unroll
        for (int m = 1; m < 32; m <<= 1) ssum += __shfl_xor(ssum, m, 32);
        const float r = rsqrtf(ssum * (1.0f / 128.0f) + EPSV);
        const float* scl = (t3 == 0) ? qs : ks;
        x0 *= r * scl[d0 + 0]; x1 *= r * scl[d0 + 1];
        x2 *= r * scl[d0 + 2]; x3 *= r * scl[d0 + 3];
        const float* fr = rope + ((size_t)l * 64 + 2 * lane) * 4;
        float y0 = fr[0] * x0 + fr[1] * x1;
        float y1 = fr[2] * x0 + fr[3] * x1;
        float y2 = fr[4] * x2 + fr[5] * x3;
        float y3 = fr[6] * x2 + fr[7] * x3;
        __bf16* dst = ((t3 == 0) ? qb : kb) + (((size_t)(b * NHEAD + h) * SEQ + l) * DH + d0);
        dst[0] = (__bf16)y0; dst[1] = (__bf16)y1; dst[2] = (__bf16)y2; dst[3] = (__bf16)y3;
    }
}

// ---------------- flash attention: per-wave 16 queries ----------------
__global__ __launch_bounds__(256) void attn_kernel(const __bf16* __restrict__ Q,
                                                   const __bf16* __restrict__ K,
                                                   const __bf16* __restrict__ VT,
                                                   __bf16* __restrict__ O) {
    __shared__ __align__(64) __bf16 pbuf[8][16 * 32];
    const int lane = threadIdx.x & 31, wid = threadIdx.x >> 5;
    const int bh = blockIdx.y;
    const int b = bh / NHEAD, h = bh % NHEAD;
    const int q0 = blockIdx.x * 128 + wid * 16;
    const __bf16* Qb = Q + (size_t)bh * SEQ * DH;
    const __bf16* Kb = K + (size_t)bh * SEQ * DH;
    const __bf16* Vb = VT + (size_t)bh * DH * SEQ;
    v16bf aq[4];
#pragma unroll
    for (int c = 0; c < 4; c++) aq[c] = load_a_frag(Qb + (size_t)q0 * DH + c * 32, DH, lane);
    v8f o[8];
    float m[8], lsum[8];
#pragma unroll
    for (int r = 0; r < 8; r++) { m[r] = -1e30f; lsum[r] = 0.f; }
#pragma unroll
    for (int j = 0; j < 8; j++)
#pragma unroll
        for (int r = 0; r < 8; r++) o[j][r] = 0.f;
    const float sscale = 0.08838834764831845f;   // 1/sqrt(128)

    for (int kbs = 0; kbs < SEQ; kbs += 32) {
        v8f s[2];
#pragma unroll
        for (int n = 0; n < 2; n++) {
#pragma unroll
            for (int r = 0; r < 8; r++) s[n][r] = 0.f;
#pragma unroll
            for (int c = 0; c < 4; c++) {
                v16bf bk = load_b_frag(Kb + (size_t)(kbs + n * 16) * DH + c * 32, DH, lane);
                s[n] = wmma_bf16(aq[c], bk, s[n]);
            }
        }
#pragma unroll
        for (int r = 0; r < 8; r++) {
            float v0 = s[0][r] * sscale, v1 = s[1][r] * sscale;
            float mx = fmaxf(v0, v1);
            mx = fmaxf(mx, __shfl_xor(mx, 1, 32));
            mx = fmaxf(mx, __shfl_xor(mx, 2, 32));
            mx = fmaxf(mx, __shfl_xor(mx, 4, 32));
            mx = fmaxf(mx, __shfl_xor(mx, 8, 32));
            float mnew = fmaxf(m[r], mx);
            float alpha = __expf(m[r] - mnew);
            float p0 = __expf(v0 - mnew), p1 = __expf(v1 - mnew);
            float rs = p0 + p1;
            rs += __shfl_xor(rs, 1, 32);
            rs += __shfl_xor(rs, 2, 32);
            rs += __shfl_xor(rs, 4, 32);
            rs += __shfl_xor(rs, 8, 32);
            lsum[r] = lsum[r] * alpha + rs;
            m[r] = mnew;
#pragma unroll
            for (int j = 0; j < 8; j++) o[j][r] *= alpha;
            int mrow = r + 8 * (lane >> 4);
            pbuf[wid][mrow * 32 + (lane & 15)] = (__bf16)p0;
            pbuf[wid][mrow * 32 + 16 + (lane & 15)] = (__bf16)p1;
        }
        v16bf ap = load_a_frag(&pbuf[wid][0], 32, lane);
#pragma unroll
        for (int j = 0; j < 8; j++) {
            v16bf bv = load_b_frag(Vb + (size_t)(j * 16) * SEQ + kbs, SEQ, lane);
            o[j] = wmma_bf16(ap, bv, o[j]);
        }
    }
#pragma unroll
    for (int r = 0; r < 8; r++) {
        float inv = 1.0f / lsum[r];
        int mrow = r + 8 * (lane >> 4);
        size_t orow = ((size_t)b * SEQ + (q0 + mrow)) * (NHEAD * DH) + h * DH;
#pragma unroll
        for (int j = 0; j < 8; j++)
            O[orow + j * 16 + (lane & 15)] = (__bf16)(o[j][r] * inv);
    }
}

// ---------------- GEMM3: [attn|gelu(mlp)](8192x15360) @ w2t^T + residual/gate ----------------
__global__ __launch_bounds__(256) void gemm3_kernel(const __bf16* __restrict__ ATT,
                                                    const __bf16* __restrict__ MLPB,
                                                    const __bf16* __restrict__ WT,
                                                    const float* __restrict__ b2,
                                                    const float* __restrict__ emb,
                                                    const float* __restrict__ hs,
                                                    float* __restrict__ out) {
    __shared__ __align__(64) __bf16 As[2][128 * 64];
    __shared__ __align__(64) __bf16 Bs[2][128 * 64];
    const int tid = threadIdx.x, lane = tid & 31, wid = tid >> 5;
    const int wm = wid >> 1, wn = wid & 1;
    const int row0 = blockIdx.y * 128, col0 = blockIdx.x * 128;
    v8f acc[2][4];
#pragma unroll
    for (int i = 0; i < 2; i++)
#pragma unroll
        for (int j = 0; j < 4; j++)
#pragma unroll
            for (int r = 0; r < 8; r++) acc[i][j][r] = 0.f;

    const int nkt = KTOT2 / 64;  // 240

    // issue tile 0 (k0 = 0 < DIM -> ATT source)
#pragma unroll
    for (int t = 0; t < 4; t++) {
        int ci = tid + t * 256;
        int rr = ci >> 3;
        int cc = (ci & 7) * 8;
        async_copy16(&ATT[(size_t)(row0 + rr) * DIM + cc], &As[0][rr * 64 + cc]);
        async_copy16(&WT[(size_t)(col0 + rr) * KTOT2 + cc], &Bs[0][rr * 64 + cc]);
    }

    for (int kt = 0; kt < nkt; kt++) {
        const int cur = kt & 1;
        if (kt + 1 < nkt) {
            const int nxt = cur ^ 1;
            const int k0 = (kt + 1) * 64;
            const __bf16* Asrc; size_t astride; int kk;
            if (k0 < DIM) { Asrc = ATT;  astride = DIM;  kk = k0; }
            else          { Asrc = MLPB; astride = MLPD; kk = k0 - DIM; }
#pragma unroll
            for (int t = 0; t < 4; t++) {
                int ci = tid + t * 256;
                int rr = ci >> 3;
                int cc = (ci & 7) * 8;
                async_copy16(&Asrc[(size_t)(row0 + rr) * astride + kk + cc], &As[nxt][rr * 64 + cc]);
                async_copy16(&WT[(size_t)(col0 + rr) * KTOT2 + k0 + cc], &Bs[nxt][rr * 64 + cc]);
            }
            wait_async_prev_tile();
        } else {
            wait_async_all();
        }
        __syncthreads();
#pragma unroll
        for (int kc = 0; kc < 2; kc++) {
            v16bf a[2], bb[4];
#pragma unroll
            for (int i = 0; i < 2; i++)
                a[i] = load_a_frag(&As[cur][(wm * 32 + i * 16) * 64 + kc * 32], 64, lane);
#pragma unroll
            for (int j = 0; j < 4; j++)
                bb[j] = load_b_frag(&Bs[cur][(wn * 64 + j * 16) * 64 + kc * 32], 64, lane);
#pragma unroll
            for (int i = 0; i < 2; i++)
#pragma unroll
                for (int j = 0; j < 4; j++)
                    acc[i][j] = wmma_bf16(a[i], bb[j], acc[i][j]);
        }
        __syncthreads();
    }
#pragma unroll
    for (int i = 0; i < 2; i++)
#pragma unroll
        for (int j = 0; j < 4; j++)
#pragma unroll
            for (int r = 0; r < 8; r++) {
                int mm = wm * 32 + i * 16 + r + 8 * (lane >> 4);
                int nn = wn * 64 + j * 16 + (lane & 15);
                int rg = row0 + mm, cg = col0 + nn;
                float hval = acc[i][j][r] + b2[cg];
                int bidx = rg >> 12;
                float g = emb[(size_t)bidx * NQKV + 2 * DIM + cg];
                out[(size_t)rg * DIM + cg] = hs[(size_t)rg * DIM + cg] + g * hval;
            }
}

// ---------------- launch ----------------
extern "C" void kernel_launch(void* const* d_in, const int* in_sizes, int n_in,
                              void* d_out, int out_size, void* d_ws, size_t ws_size,
                              hipStream_t stream) {
    const float* hs     = (const float*)d_in[0];
    const float* temb   = (const float*)d_in[1];
    const float* rope   = (const float*)d_in[2];
    const float* w_lin  = (const float*)d_in[3];
    const float* b_lin  = (const float*)d_in[4];
    const float* w1     = (const float*)d_in[5];
    const float* b1     = (const float*)d_in[6];
    const float* w2     = (const float*)d_in[7];
    const float* b2     = (const float*)d_in[8];
    const float* qscale = (const float*)d_in[9];
    const float* kscale = (const float*)d_in[10];
    float* out = (float*)d_out;

    char* ws = (char*)d_ws;
    float*  emb   = (float*)(ws + OFF_EMB);
    __bf16* xbf   = (__bf16*)(ws + OFF_X);
    __bf16* w1t   = (__bf16*)(ws + OFF_W1T);
    __bf16* w2t   = (__bf16*)(ws + OFF_W2T);
    __bf16* qkvbf = (__bf16*)(ws + OFF_QKV);
    __bf16* mlpbf = (__bf16*)(ws + OFF_MLP);
    __bf16* qbf   = (__bf16*)(ws + OFF_Q);
    __bf16* kbf   = (__bf16*)(ws + OFF_K);
    __bf16* vtbf  = (__bf16*)(ws + OFF_VT);
    __bf16* attbf = (__bf16*)(ws + OFF_ATT);

    convert_w1t_kernel<<<8192, 256, 0, stream>>>(w1, w1t);
    convert_w2t_kernel<<<8192, 256, 0, stream>>>(w2, w2t);
    emb_kernel<<<dim3(NQKV / 256, BATCH), 256, 0, stream>>>(temb, w_lin, b_lin, emb);
    lnmod_kernel<<<NROWS, 256, 0, stream>>>(hs, emb, xbf);
    gemm1_kernel<<<dim3(N1 / 128, NROWS / 128), 256, 0, stream>>>(xbf, w1t, b1, qkvbf, mlpbf);
    rmsrope_kernel<<<(BATCH * SEQ * 3 * NHEAD) / 8, 256, 0, stream>>>(qkvbf, rope, qscale, kscale,
                                                                     qbf, kbf, vtbf);
    attn_kernel<<<dim3(SEQ / 128, BATCH * NHEAD), 256, 0, stream>>>(qbf, kbf, vtbf, attbf);
    gemm3_kernel<<<dim3(DIM / 128, NROWS / 128), 256, 0, stream>>>(attbf, mlpbf, w2t, b2, emb, hs, out);
}
